// Mamba2Block_69767448756296
// MI455X (gfx1250) — matrix operand order repaired and checked by reading, not compiled, not run on hardware
//
#include <hip/hip_runtime.h>

// ---------------- types ----------------
typedef __attribute__((ext_vector_type(16))) __bf16 v16bf;
typedef __attribute__((ext_vector_type(8)))  __bf16 v8bf;
typedef __attribute__((ext_vector_type(8)))  float  v8f;
typedef int async_v4i __attribute__((vector_size(16)));

#define B_SZ 2
#define L_SZ 2048
#define DM   2048
#define DIN  4096
#define HEADS 64
#define HD   64
#define STATE 128
#define CONV_DIM 4352
#define PROJ 8512
#define CHUNK 256
#define NC   8
#define KCONV 4
#define EPSN 1e-5f
#define LDT 48   // padded LDS tile stride (elements): 96B, 16B-aligned rows

// CDNA5 async global->LDS copy (ASYNCcnt), with sync fallback.
#if defined(__HIP_DEVICE_COMPILE__) && \
    __has_builtin(__builtin_amdgcn_global_load_async_to_lds_b128)
#define HAS_ASYNC_LDS 1
#else
#define HAS_ASYNC_LDS 0
#endif

#if HAS_ASYNC_LDS
#define ASYNC_CP16(gp, lp)                                                     \
  __builtin_amdgcn_global_load_async_to_lds_b128(                              \
      (__attribute__((address_space(1))) async_v4i*)(gp),                      \
      (__attribute__((address_space(3))) async_v4i*)(lp), 0, 0)
#endif

static __device__ inline void wait_async0() {
#if defined(__HIP_DEVICE_COMPILE__)
#if __has_builtin(__builtin_amdgcn_s_wait_asynccnt)
  __builtin_amdgcn_s_wait_asynccnt(0);
#else
  asm volatile("s_wait_asynccnt 0x0" ::: "memory");
#endif
#endif
}

static __device__ inline v8f v8f_zero() {
  v8f v;
#pragma unroll
  for (int i = 0; i < 8; ++i) v[i] = 0.f;
  return v;
}

union FragU { v16bf v; struct { v8bf lo, hi; } h; };

// A-fragment (16x32 bf16), ISA layout: lanes 0-15 row m hold K {0-7,16-23},
// lanes 16-31 hold K {8-15,24-31}. tile is row-major [rows][LDT].
static __device__ inline v16bf load_a_frag(const __bf16* tile, int row, int ld) {
  unsigned lane = threadIdx.x & 31u;
  unsigned half = lane >> 4;
  unsigned m = lane & 15u;
  const __bf16* p = tile + (size_t)(row + m) * ld + half * 8;
  FragU f;
  f.h.lo = *(const v8bf*)(p);
  f.h.hi = *(const v8bf*)(p + 16);
  return f.v;
}

// B-fragment (32x16 bf16): lane = column n (lane&15); lanes 0-15 hold K {0-7,16-23},
// lanes 16-31 hold K {8-15,24-31}. tile stored as [N][K] row-major with stride ld.
static __device__ inline v16bf load_b_frag(const __bf16* tile, int col, int ld) {
  unsigned lane = threadIdx.x & 31u;
  unsigned kh = lane >> 4;
  unsigned n = lane & 15u;
  const __bf16* p = tile + (size_t)(col + n) * ld + kh * 8;
  FragU f;
  f.h.lo = *(const v8bf*)(p);
  f.h.hi = *(const v8bf*)(p + 16);
  return f.v;
}

static __device__ inline v8f wmma_bf16(v16bf a, v16bf b, v8f c) {
  return __builtin_amdgcn_wmma_f32_16x16x32_bf16(false, a, false, b, (short)0, c,
                                                 false, false);
}

// ---------------- f32 -> bf16 convert ----------------
__global__ void cvt_bf16(const float* __restrict__ src, __bf16* __restrict__ dst,
                         long long n) {
  long long i = (long long)blockIdx.x * blockDim.x + threadIdx.x;
  if (i < n) dst[i] = (__bf16)src[i];
}

// ---------------- generic batched GEMM: C = A @ W^T ----------------
// A: [M][K] bf16 row-major, W: [N][K] bf16 row-major, C: [M][N] f32.
// K must be a multiple of 32. Strides are per-batch element counts.
// Double-buffered LDS pipeline; global->LDS via CDNA5 async copies when
// available (ASYNCcnt), else VGPR-staged loads.
__global__ void __launch_bounds__(256)
gemm_bf16(const __bf16* __restrict__ A, const __bf16* __restrict__ W,
          float* __restrict__ C, int M, int N, int Kd,
          long long sA, long long sW, long long sC) {
  A += (size_t)blockIdx.z * sA;
  W += (size_t)blockIdx.z * sW;
  C += (size_t)blockIdx.z * sC;
  const int m0 = blockIdx.y * 128, n0 = blockIdx.x * 128;
  __shared__ __align__(16) __bf16 As[2][128 * LDT];
  __shared__ __align__(16) __bf16 Bs[2][128 * LDT];
  const int tid = threadIdx.x;
  const int wave = tid >> 5;
  const int wm = (wave >> 2) * 64;  // 0 or 64
  const int wn = (wave & 3) * 32;   // 0..96
  v8f acc[4][2];
#pragma unroll
  for (int i = 0; i < 4; ++i)
#pragma unroll
    for (int j = 0; j < 2; ++j) acc[i][j] = v8f_zero();

  const int lr = tid >> 1;        // 0..127
  const int lc = (tid & 1) * 16;  // 0 or 16
  // Clamp ragged rows: duplicated data lands in columns discarded at store.
  const int gm = (m0 + lr < M) ? (m0 + lr) : (M - 1);
  const int gn = (n0 + lr < N) ? (n0 + lr) : (N - 1);
  const __bf16* ga = A + (size_t)gm * Kd + lc;
  const __bf16* gb = W + (size_t)gn * Kd + lc;
  __bf16* la = &As[0][0] + lr * LDT + lc;
  __bf16* lb = &Bs[0][0] + lr * LDT + lc;
  const size_t bufstride = 128 * LDT;

  auto issue = [&](int k0, int bi) {
#if HAS_ASYNC_LDS
    ASYNC_CP16(ga + k0,     la + (size_t)bi * bufstride);
    ASYNC_CP16(ga + k0 + 8, la + (size_t)bi * bufstride + 8);
    ASYNC_CP16(gb + k0,     lb + (size_t)bi * bufstride);
    ASYNC_CP16(gb + k0 + 8, lb + (size_t)bi * bufstride + 8);
#else
    v8bf a0 = *(const v8bf*)(ga + k0);
    v8bf a1 = *(const v8bf*)(ga + k0 + 8);
    v8bf b0 = *(const v8bf*)(gb + k0);
    v8bf b1 = *(const v8bf*)(gb + k0 + 8);
    *(v8bf*)(la + (size_t)bi * bufstride) = a0;
    *(v8bf*)(la + (size_t)bi * bufstride + 8) = a1;
    *(v8bf*)(lb + (size_t)bi * bufstride) = b0;
    *(v8bf*)(lb + (size_t)bi * bufstride + 8) = b1;
#endif
  };

  const int ntiles = Kd >> 5;
  issue(0, 0);
  wait_async0();
  __syncthreads();
  for (int t = 0; t < ntiles; ++t) {
    const int cur = t & 1;
    if (t + 1 < ntiles) issue((t + 1) << 5, cur ^ 1);
    v16bf af[4], bfr[2];
#pragma unroll
    for (int i = 0; i < 4; ++i) af[i] = load_a_frag(As[cur], wm + i * 16, LDT);
#pragma unroll
    for (int j = 0; j < 2; ++j) bfr[j] = load_b_frag(Bs[cur], wn + j * 16, LDT);
#pragma unroll
    for (int i = 0; i < 4; ++i)
#pragma unroll
      for (int j = 0; j < 2; ++j) acc[i][j] = wmma_bf16(af[i], bfr[j], acc[i][j]);
    wait_async0();
    __syncthreads();
  }
  unsigned lane = tid & 31u;
  unsigned nn = lane & 15u;
  unsigned mh = (lane >> 4) * 8;
#pragma unroll
  for (int i = 0; i < 4; ++i)
#pragma unroll
    for (int j = 0; j < 2; ++j) {
      int col = n0 + wn + j * 16 + nn;
      if (col >= N) continue;
      int mb = m0 + wm + i * 16 + mh;
#pragma unroll
      for (int r = 0; r < 8; ++r) {
        int row = mb + r;
        if (row < M) C[(size_t)row * N + col] = acc[i][j][r];
      }
    }
}

// ---------------- conv1d(K=4) + SiLU + split + softplus ----------------
__global__ void __launch_bounds__(256)
conv_kernel(const float* __restrict__ proj, const float* __restrict__ conv_w,
            const float* __restrict__ conv_b, const float* __restrict__ dt_bias,
            const float* __restrict__ A_log, __bf16* __restrict__ xb,
            __bf16* __restrict__ xdtb, __bf16* __restrict__ Bb,
            __bf16* __restrict__ Cb, float* __restrict__ dAbuf) {
  long long g = (long long)blockIdx.x * blockDim.x + threadIdx.x;
  if (g >= (long long)B_SZ * L_SZ * CONV_DIM) return;
  int ch = (int)(g % CONV_DIM);
  long long bl = g / CONV_DIM;
  int l = (int)(bl % L_SZ);
  long long bbase = bl - l;  // b*L
  float acc = conv_b[ch];
#pragma unroll
  for (int i = 0; i < KCONV; ++i) {
    int li = l - (KCONV - 1) + i;
    if (li >= 0) acc += conv_w[ch * KCONV + i] * proj[(bbase + li) * PROJ + DIN + ch];
  }
  float s = acc / (1.f + __expf(-acc));  // SiLU
  if (ch < DIN) {
    int h = ch >> 6;  // HD = 64
    float dtr = proj[bl * PROJ + DIN + CONV_DIM + h] + dt_bias[h];
    float dt = (dtr > 20.f) ? dtr : log1pf(__expf(dtr));  // softplus
    xb[bl * DIN + ch] = (__bf16)s;
    xdtb[bl * DIN + ch] = (__bf16)(s * dt);
    if ((ch & 63) == 0) dAbuf[bl * HEADS + h] = -dt * __expf(A_log[h]);
  } else if (ch < DIN + STATE) {
    Bb[bl * STATE + (ch - DIN)] = (__bf16)s;
  } else {
    Cb[bl * STATE + (ch - DIN - STATE)] = (__bf16)s;
  }
}

// ---------------- inclusive cumsum of dA within each chunk ----------------
__global__ void __launch_bounds__(256)
scan_kernel(const float* __restrict__ dAbuf, float* __restrict__ dAcs) {
  int idx = blockIdx.x;
  int c = idx % NC;
  int h = (idx / NC) % HEADS;
  int b = idx / (NC * HEADS);
  int t = threadIdx.x;
  __shared__ float sh[CHUNK];
  sh[t] = dAbuf[((long long)(b * L_SZ) + c * CHUNK + t) * HEADS + h];
  __syncthreads();
  for (int off = 1; off < CHUNK; off <<= 1) {
    float v = (t >= off) ? sh[t - off] : 0.f;
    __syncthreads();
    sh[t] += v;
    __syncthreads();
  }
  dAcs[((long long)(b * HEADS + h) * NC + c) * CHUNK + t] = sh[t];
}

// ---------------- Y_diag = (L o CB) @ Xdt per (b,c,h) ----------------
__global__ void __launch_bounds__(256)
ydiag_kernel(const float* __restrict__ CB, const float* __restrict__ dAcs,
             const __bf16* __restrict__ xdtb, float* __restrict__ Y) {
  const int h = blockIdx.x;
  const int bc = blockIdx.y;
  const int b = bc / NC, c = bc % NC;
  const int tid = threadIdx.x;
  const float* cb = CB + (size_t)bc * CHUNK * CHUNK;
  const float* da = dAcs + ((size_t)(b * HEADS + h) * NC + c) * CHUNK;
  __shared__ float dal[CHUNK];
  __shared__ __align__(16) __bf16 Ms[CHUNK * LDT];
  __shared__ __align__(16) __bf16 Xs[64 * LDT];
  dal[tid] = da[tid];
  __syncthreads();
  const int wave = tid >> 5;
  const int mtb = wave * 32;  // two 16-row tiles per wave
  v8f acc[2][4];
#pragma unroll
  for (int i = 0; i < 2; ++i)
#pragma unroll
    for (int j = 0; j < 4; ++j) acc[i][j] = v8f_zero();
  const long long lbase = (long long)b * L_SZ + c * CHUNK;
  for (int s0 = 0; s0 < CHUNK; s0 += 32) {
    {  // masked decay matrix tile, row t = tid
      float dt_ = dal[tid];
      const float* crow = cb + (size_t)tid * CHUNK + s0;
#pragma unroll 8
      for (int j = 0; j < 32; ++j) {
        int s = s0 + j;
        float v = (tid >= s) ? crow[j] * __expf(dt_ - dal[s]) : 0.f;
        Ms[tid * LDT + j] = (__bf16)v;
      }
    }
    {  // Xs[p][s] = xdt[s][p]
      int p = tid >> 2;
      int jb = (tid & 3) * 8;
#pragma unroll
      for (int j = 0; j < 8; ++j) {
        int s = jb + j;
        Xs[p * LDT + s] = xdtb[(lbase + s0 + s) * DIN + h * HD + p];
      }
    }
    __syncthreads();
    v16bf af[2], bfr[4];
#pragma unroll
    for (int i = 0; i < 2; ++i) af[i] = load_a_frag(Ms, mtb + i * 16, LDT);
#pragma unroll
    for (int j = 0; j < 4; ++j) bfr[j] = load_b_frag(Xs, j * 16, LDT);
#pragma unroll
    for (int i = 0; i < 2; ++i)
#pragma unroll
      for (int j = 0; j < 4; ++j) acc[i][j] = wmma_bf16(af[i], bfr[j], acc[i][j]);
    __syncthreads();
  }
  unsigned lane = tid & 31u;
  unsigned nn = lane & 15u, mh = (lane >> 4) * 8;
#pragma unroll
  for (int i = 0; i < 2; ++i)
#pragma unroll
    for (int j = 0; j < 4; ++j) {
      int p = j * 16 + nn;
      int mb = mtb + i * 16 + mh;
#pragma unroll
      for (int r = 0; r < 8; ++r) {
        int row = mb + r;
        Y[(lbase + row) * DIN + h * HD + p] = acc[i][j][r];
      }
    }
}

// ---------------- chunk states = Xdt^T @ (B * decay) per (b,c,h) ----------------
__global__ void __launch_bounds__(256)
states_kernel(const __bf16* __restrict__ Bb, const __bf16* __restrict__ xdtb,
              const float* __restrict__ dAcs, float* __restrict__ states) {
  const int h = blockIdx.x;
  const int bc = blockIdx.y;
  const int b = bc / NC, c = bc % NC;
  const int tid = threadIdx.x;
  const float* da = dAcs + ((size_t)(b * HEADS + h) * NC + c) * CHUNK;
  __shared__ float dal[CHUNK];
  __shared__ __align__(16) __bf16 XT[64 * LDT];
  __shared__ __align__(16) __bf16 Bd[128 * LDT];
  dal[tid] = da[tid];
  __syncthreads();
  const float dlast = dal[CHUNK - 1];
  const int wave = tid >> 5;
  const int mt = (wave >> 1) * 16;   // p tile base: 0,16,32,48
  const int ntb = (wave & 1) * 64;   // n half: 0 or 64
  v8f acc[4];
#pragma unroll
  for (int j = 0; j < 4; ++j) acc[j] = v8f_zero();
  const long long lbase = (long long)b * L_SZ + c * CHUNK;
  for (int s0 = 0; s0 < CHUNK; s0 += 32) {
    {  // XT[p][s] = xdt[s][p]
      int p = tid >> 2;
      int jb = (tid & 3) * 8;
#pragma unroll
      for (int j = 0; j < 8; ++j) {
        int s = jb + j;
        XT[p * LDT + s] = xdtb[(lbase + s0 + s) * DIN + h * HD + p];
      }
    }
    {  // Bd[n][s] = B[s][n] * exp(dA_last - dA_cum[s])
      int n = tid >> 1;
      int jb = (tid & 1) * 16;
#pragma unroll
      for (int j = 0; j < 16; ++j) {
        int s = jb + j;
        float dec = __expf(dlast - dal[s0 + s]);
        float v = (float)Bb[(lbase + s0 + s) * STATE + n] * dec;
        Bd[n * LDT + s] = (__bf16)v;
      }
    }
    __syncthreads();
    v16bf a = load_a_frag(XT, mt, LDT);
#pragma unroll
    for (int j = 0; j < 4; ++j) {
      v16bf bfr = load_b_frag(Bd, ntb + j * 16, LDT);
      acc[j] = wmma_bf16(a, bfr, acc[j]);
    }
    __syncthreads();
  }
  unsigned lane = tid & 31u;
  unsigned nn = lane & 15u, mh = (lane >> 4) * 8;
  float* S = states + (((size_t)(b * NC + c) * HEADS + h) * HD) * STATE;
#pragma unroll
  for (int j = 0; j < 4; ++j) {
    int n = ntb + j * 16 + nn;
#pragma unroll
    for (int r = 0; r < 8; ++r) {
      int p = mt + mh + r;
      S[(size_t)p * STATE + n] = acc[j][r];
    }
  }
}

// ---------------- inter-chunk recurrence: prev_c, P = exp(d)P + S ----------------
__global__ void __launch_bounds__(256)
chain_kernel(const float* __restrict__ states, const float* __restrict__ dAcs,
             float* __restrict__ prev) {
  int bh = blockIdx.x;
  int b = bh / HEADS, h = bh % HEADS;
  int tid = threadIdx.x;
  float P[32];
#pragma unroll
  for (int i = 0; i < 32; ++i) P[i] = 0.f;
  for (int c = 0; c < NC; ++c) {
    size_t base = (((size_t)(b * NC + c) * HEADS + h) * HD) * STATE;
    float dec = __expf(dAcs[((size_t)(b * HEADS + h) * NC + c) * CHUNK + (CHUNK - 1)]);
#pragma unroll
    for (int i = 0; i < 32; ++i) {
      int idx = tid + i * 256;
      prev[base + idx] = P[i];
      P[i] = P[i] * dec + states[base + idx];
    }
  }
}

// ---------------- Y_off += exp(dA_cum[t]) * C @ prev^T per (b,c,h) ----------------
__global__ void __launch_bounds__(256)
yoff_kernel(const __bf16* __restrict__ Cb, const float* __restrict__ prev,
            const float* __restrict__ dAcs, float* __restrict__ Y) {
  const int h = blockIdx.x;
  const int bc = blockIdx.y;
  const int b = bc / NC, c = bc % NC;
  const int tid = threadIdx.x;
  const float* da = dAcs + ((size_t)(b * HEADS + h) * NC + c) * CHUNK;
  __shared__ float dal[CHUNK];
  __shared__ __align__(16) __bf16 Cs[CHUNK * LDT];
  __shared__ __align__(16) __bf16 Pw[64 * LDT];
  dal[tid] = da[tid];
  const int wave = tid >> 5;
  const int mtb = wave * 32;
  v8f acc[2][4];
#pragma unroll
  for (int i = 0; i < 2; ++i)
#pragma unroll
    for (int j = 0; j < 4; ++j) acc[i][j] = v8f_zero();
  const long long lbase = (long long)b * L_SZ + c * CHUNK;
  const float* pv = prev + (((size_t)(b * NC + c) * HEADS + h) * HD) * STATE;
  for (int k0 = 0; k0 < STATE; k0 += 32) {
    {  // Cs row t = tid, 32 n-values
      const __bf16* crow = Cb + (lbase + tid) * STATE + k0;
#pragma unroll 8
      for (int j = 0; j < 32; ++j) Cs[tid * LDT + j] = crow[j];
    }
    {  // Pw[p][n] from prev (f32 -> bf16)
      int p = tid >> 2;
      int jb = (tid & 3) * 8;
#pragma unroll
      for (int j = 0; j < 8; ++j)
        Pw[p * LDT + jb + j] = (__bf16)pv[(size_t)p * STATE + k0 + jb + j];
    }
    __syncthreads();
    v16bf af[2], bfr[4];
#pragma unroll
    for (int i = 0; i < 2; ++i) af[i] = load_a_frag(Cs, mtb + i * 16, LDT);
#pragma unroll
    for (int j = 0; j < 4; ++j) bfr[j] = load_b_frag(Pw, j * 16, LDT);
#pragma unroll
    for (int i = 0; i < 2; ++i)
#pragma unroll
      for (int j = 0; j < 4; ++j) acc[i][j] = wmma_bf16(af[i], bfr[j], acc[i][j]);
    __syncthreads();
  }
  unsigned lane = tid & 31u;
  unsigned nn = lane & 15u, mh = (lane >> 4) * 8;
#pragma unroll
  for (int i = 0; i < 2; ++i)
#pragma unroll
    for (int j = 0; j < 4; ++j) {
      int p = j * 16 + nn;
      int mb = mtb + i * 16 + mh;
#pragma unroll
      for (int r = 0; r < 8; ++r) {
        int row = mb + r;
        Y[(lbase + row) * DIN + h * HD + p] += acc[i][j][r] * __expf(dal[row]);
      }
    }
}

// ---------------- +D*x, gate SiLU, RMS norm -> bf16 ----------------
__global__ void __launch_bounds__(256)
gate_norm_kernel(const float* __restrict__ Y, const float* __restrict__ proj,
                 const __bf16* __restrict__ xb, const float* __restrict__ Dp,
                 const float* __restrict__ norm_w, __bf16* __restrict__ ynb) {
  long long bl = blockIdx.x;
  int tid = threadIdx.x;
  const float* yr = Y + bl * DIN;
  const float* gr = proj + bl * PROJ;  // gate = first DIN of proj row
  float hg[16];
  float ss = 0.f;
#pragma unroll
  for (int i = 0; i < 16; ++i) {
    int d = tid + i * 256;
    int h = d >> 6;
    float xv = (float)xb[bl * DIN + d];
    float yv = yr[d] + Dp[h] * xv;
    float g = gr[d];
    float v = yv * (g / (1.f + __expf(-g)));
    hg[i] = v;
    ss += v * v;
  }
  __shared__ float red[256];
  red[tid] = ss;
  __syncthreads();
  for (int off = 128; off > 0; off >>= 1) {
    if (tid < off) red[tid] += red[tid + off];
    __syncthreads();
  }
  float rs = rsqrtf(red[0] / (float)DIN + EPSN);
#pragma unroll
  for (int i = 0; i < 16; ++i) {
    int d = tid + i * 256;
    ynb[bl * DIN + d] = (__bf16)(hg[i] * rs * norm_w[d]);
  }
}

// ---------------- host launcher ----------------
extern "C" void kernel_launch(void* const* d_in, const int* in_sizes, int n_in,
                              void* d_out, int out_size, void* d_ws, size_t ws_size,
                              hipStream_t stream) {
  (void)in_sizes; (void)n_in; (void)out_size; (void)ws_size;
  const float* hs    = (const float*)d_in[0];
  const float* wi    = (const float*)d_in[1];
  const float* cw    = (const float*)d_in[2];
  const float* cbias = (const float*)d_in[3];
  const float* dtb   = (const float*)d_in[4];
  const float* alog  = (const float*)d_in[5];
  const float* Dp    = (const float*)d_in[6];
  const float* nw    = (const float*)d_in[7];
  const float* wo    = (const float*)d_in[8];
  float* out = (float*)d_out;

  char* p = (char*)d_ws;
  auto take = [&](size_t bytes) -> void* {
    void* q = (void*)p;
    p += (bytes + 255) & ~(size_t)255;
    return q;
  };
  __bf16* hsb  = (__bf16*)take((size_t)B_SZ * L_SZ * DM * 2);
  __bf16* wib  = (__bf16*)take((size_t)PROJ * DM * 2);
  __bf16* wob  = (__bf16*)take((size_t)DM * DIN * 2);
  float*  proj = (float*)take((size_t)B_SZ * L_SZ * PROJ * 4);
  __bf16* xb   = (__bf16*)take((size_t)B_SZ * L_SZ * DIN * 2);
  __bf16* xdtb = (__bf16*)take((size_t)B_SZ * L_SZ * DIN * 2);
  __bf16* Bb   = (__bf16*)take((size_t)B_SZ * L_SZ * STATE * 2);
  __bf16* Cbp  = (__bf16*)take((size_t)B_SZ * L_SZ * STATE * 2);
  float*  dAb  = (float*)take((size_t)B_SZ * L_SZ * HEADS * 4);
  float*  dAcs = (float*)take((size_t)B_SZ * HEADS * NC * CHUNK * 4);
  float*  CBb  = (float*)take((size_t)B_SZ * NC * CHUNK * CHUNK * 4);
  float*  st   = (float*)take((size_t)B_SZ * NC * HEADS * HD * STATE * 4);
  float*  pv   = (float*)take((size_t)B_SZ * NC * HEADS * HD * STATE * 4);
  float*  Y    = (float*)take((size_t)B_SZ * L_SZ * DIN * 4);
  __bf16* ynb  = (__bf16*)take((size_t)B_SZ * L_SZ * DIN * 2);

  long long n;
  n = (long long)B_SZ * L_SZ * DM;
  cvt_bf16<<<(unsigned)((n + 255) / 256), 256, 0, stream>>>(hs, hsb, n);
  n = (long long)PROJ * DM;
  cvt_bf16<<<(unsigned)((n + 255) / 256), 256, 0, stream>>>(wi, wib, n);
  n = (long long)DM * DIN;
  cvt_bf16<<<(unsigned)((n + 255) / 256), 256, 0, stream>>>(wo, wob, n);

  // in_proj: proj = hs @ in_proj_w^T   (4096 x 8512 x 2048)
  gemm_bf16<<<dim3((PROJ + 127) / 128, (B_SZ * L_SZ) / 128, 1), 256, 0, stream>>>(
      hsb, wib, proj, B_SZ * L_SZ, PROJ, DM, 0, 0, 0);

  n = (long long)B_SZ * L_SZ * CONV_DIM;
  conv_kernel<<<(unsigned)((n + 255) / 256), 256, 0, stream>>>(
      proj, cw, cbias, dtb, alog, xb, xdtb, Bb, Cbp, dAb);

  scan_kernel<<<B_SZ * HEADS * NC, CHUNK, 0, stream>>>(dAb, dAcs);

  // per-chunk CB = C_chunk @ B_chunk^T (batch = B*NC, 256x256x128)
  gemm_bf16<<<dim3(2, 2, B_SZ * NC), 256, 0, stream>>>(
      Cbp, Bb, CBb, CHUNK, CHUNK, STATE, (long long)CHUNK * STATE,
      (long long)CHUNK * STATE, (long long)CHUNK * CHUNK);

  ydiag_kernel<<<dim3(HEADS, B_SZ * NC), 256, 0, stream>>>(CBb, dAcs, xdtb, Y);
  states_kernel<<<dim3(HEADS, B_SZ * NC), 256, 0, stream>>>(Bb, xdtb, dAcs, st);
  chain_kernel<<<B_SZ * HEADS, 256, 0, stream>>>(st, dAcs, pv);
  yoff_kernel<<<dim3(HEADS, B_SZ * NC), 256, 0, stream>>>(Cbp, pv, dAcs, Y);
  gate_norm_kernel<<<B_SZ * L_SZ, 256, 0, stream>>>(Y, proj, xb, Dp, nw, ynb);

  // out_proj: out = ynorm @ out_proj_w^T  (4096 x 2048 x 4096)
  gemm_bf16<<<dim3(DM / 128, (B_SZ * L_SZ) / 128, 1), 256, 0, stream>>>(
      ynb, wob, out, B_SZ * L_SZ, DM, DIN, 0, 0, 0);
}
